// PointSAModuleMSG_3143916060896
// MI455X (gfx1250) — matrix operand order repaired and checked by reading, not compiled, not run on hardware
//
#include <hip/hip_runtime.h>

// ---------------------------------------------------------------------------
// Problem constants (from the reference)
// ---------------------------------------------------------------------------
#define B_   16
#define N_   8192
#define M_   1024
#define CF_  96
#define COUT_TOTAL 256   // 128 + 128 concatenated channels

typedef __attribute__((ext_vector_type(16))) __bf16 v16bf;
typedef __attribute__((ext_vector_type(8)))  float  v8f;

union ABFrag { uint4 u[2]; v16bf v; };

__device__ __forceinline__ unsigned short f2bf(float f) {
    union { float f; unsigned u; } v; v.f = f;
    unsigned r = v.u + 0x7fffu + ((v.u >> 16) & 1u);   // round-to-nearest-even
    return (unsigned short)(r >> 16);
}
__device__ __forceinline__ unsigned pack2bf(float a, float b) {
    return (unsigned)f2bf(a) | ((unsigned)f2bf(b) << 16);
}

// A fragment: 16x32 bf16, row-major LDS (ld = Kp elements).
// lane L: m = rt*16 + L%16 ; half = L/16 ; elems 0-7 -> k = kk+8*half..+7,
// elems 8-15 -> k = kk+16+8*half..+7  (ISA 7.12.2)
__device__ __forceinline__ v16bf loadA(const unsigned short* w, int Kp, int rt, int kk, int lane) {
    int half = lane >> 4;
    int m = rt * 16 + (lane & 15);
    const unsigned short* base = w + m * Kp + kk + half * 8;
    ABFrag t;
    t.u[0] = *(const uint4*)(base);
    t.u[1] = *(const uint4*)(base + 16);
    return t.v;
}

// B fragment: 32x16 bf16, col-major LDS (ld = Kp elements).
// lane L: n = ct*16 + L%16 ; half = L/16 ; elems 0-7 -> k = kk+8*half..+7, elems 8-15 -> k = kk+16+8*half..+7
__device__ __forceinline__ v16bf loadB(const unsigned short* x, int Kp, int ct, int kk, int lane) {
    int half = lane >> 4;
    int n = ct * 16 + (lane & 15);
    const unsigned short* base = x + n * Kp + kk + half * 8;
    ABFrag t;
    t.u[0] = *(const uint4*)(base);
    t.u[1] = *(const uint4*)(base + 16);
    return t.v;
}

// ---------------------------------------------------------------------------
// 0. Coalesced layout transpose: (B, CF, N) -> (B, N, CF)
//    Makes later random-point gathers contiguous (384 B per point).
// ---------------------------------------------------------------------------
__global__ __launch_bounds__(256) void transpose_kernel(const float* __restrict__ feats,
                                                        float* __restrict__ featsT) {
    __shared__ float tile[32][33];
    const int tx = threadIdx.x & 31, ty = threadIdx.x >> 5;   // 32 x 8
    const int tilesP = N_ / 32;                               // 256
    const int chTiles = CF_ / 32;                             // 3
    const int b   = blockIdx.x / (tilesP * chTiles);
    const int rem = blockIdx.x % (tilesP * chTiles);
    const int ch0 = (rem / tilesP) * 32;
    const int p0  = (rem % tilesP) * 32;

    for (int i = ty; i < 32; i += 8)
        tile[i][tx] = feats[((size_t)b * CF_ + ch0 + i) * N_ + p0 + tx];
    __syncthreads();
    for (int i = ty; i < 32; i += 8)
        featsT[((size_t)b * N_ + p0 + i) * CF_ + ch0 + tx] = tile[tx][i];
}

// ---------------------------------------------------------------------------
// 1. Furthest point sampling: one workgroup per batch, cloud in LDS.
// ---------------------------------------------------------------------------
__global__ __launch_bounds__(1024) void fps_kernel(const float* __restrict__ xyz,
                                                   int* __restrict__ ws_idx,
                                                   float* __restrict__ d_out) {
    extern __shared__ char smem[];
    float* px = (float*)smem;
    float* py = px + N_;
    float* pz = py + N_;
    float* redv = pz + N_;
    int*   redi = (int*)(redv + 32);
    int*   selp = redi + 32;

    const int b = blockIdx.x;
    const int tid = threadIdx.x;
    const int wave = tid >> 5, lane = tid & 31;
    const float* base = xyz + (size_t)b * N_ * 3;

    for (int i = tid; i < N_; i += 1024) {
        px[i] = base[i * 3 + 0];
        py[i] = base[i * 3 + 1];
        pz[i] = base[i * 3 + 2];
    }
    float d[8];
#pragma unroll
    for (int j = 0; j < 8; ++j) d[j] = 1e10f;
    __syncthreads();

    float* out_new = d_out;                                           // (B,M,3)
    float* out_idx = d_out + (size_t)B_ * M_ * 3 + (size_t)B_ * COUT_TOTAL * M_;

    int sel = 0;
    for (int s = 0; s < M_; ++s) {
        if (tid == 0) {
            ws_idx[b * M_ + s] = sel;
            out_idx[b * M_ + s] = (float)sel;
            out_new[(size_t)(b * M_ + s) * 3 + 0] = px[sel];
            out_new[(size_t)(b * M_ + s) * 3 + 1] = py[sel];
            out_new[(size_t)(b * M_ + s) * 3 + 2] = pz[sel];
        }
        const float cx = px[sel], cy = py[sel], cz = pz[sel];
        float best = -1.0f; int bidx = 0;
#pragma unroll
        for (int j = 0; j < 8; ++j) {
            const int i = tid + j * 1024;
            const float dx = px[i] - cx, dy = py[i] - cy, dz = pz[i] - cz;
            const float dd = dx * dx + dy * dy + dz * dz;
            d[j] = fminf(d[j], dd);
            if (d[j] > best || (d[j] == best && i < bidx)) { best = d[j]; bidx = i; }
        }
        // wave32 butterfly argmax (lowest index wins ties -> matches jnp.argmax)
#pragma unroll
        for (int off = 16; off > 0; off >>= 1) {
            const float ov = __shfl_xor(best, off, 32);
            const int   oi = __shfl_xor(bidx, off, 32);
            if (ov > best || (ov == best && oi < bidx)) { best = ov; bidx = oi; }
        }
        if (lane == 0) { redv[wave] = best; redi[wave] = bidx; }
        __syncthreads();
        if (tid < 32) {
            best = redv[tid]; bidx = redi[tid];
#pragma unroll
            for (int off = 16; off > 0; off >>= 1) {
                const float ov = __shfl_xor(best, off, 32);
                const int   oi = __shfl_xor(bidx, off, 32);
                if (ov > best || (ov == best && oi < bidx)) { best = ov; bidx = oi; }
            }
            if (tid == 0) *selp = bidx;
        }
        __syncthreads();
        sel = *selp;
    }
}

// ---------------------------------------------------------------------------
// 2. Ball query: one wave per center, ordered first-k collection via ballot.
// ---------------------------------------------------------------------------
__global__ __launch_bounds__(256) void ballquery_kernel(const float* __restrict__ xyz,
                                                        const float* __restrict__ newxyz,
                                                        int* __restrict__ gi,
                                                        float r2, int k) {
    const int g = blockIdx.x * 8 + (threadIdx.x >> 5);
    const int lane = threadIdx.x & 31;
    const int b = g / M_, m = g % M_;

    const float* c = newxyz + (size_t)(b * M_ + m) * 3;
    const float cx = c[0], cy = c[1], cz = c[2];
    const float cn = cx * cx + cy * cy + cz * cz;
    const float* pb = xyz + (size_t)b * N_ * 3;
    int* out = gi + (size_t)(b * M_ + m) * k;

    int total = 0;
    int firstIdx = N_ - 1;
    bool haveFirst = false;

    for (int i0 = 0; i0 < N_ && total < k; i0 += 32) {
        const int p = i0 + lane;
        __builtin_prefetch(pb + (size_t)(i0 + 64) * 3, 0, 1);   // global_prefetch_b8
        const float x = pb[p * 3 + 0], y = pb[p * 3 + 1], z = pb[p * 3 + 2];
        const float pn = x * x + y * y + z * z;
        // same formulation as reference: |c|^2 + |p|^2 - 2 c.p
        const float d2 = cn + pn - 2.0f * (cx * x + cy * y + cz * z);
        const bool inball = d2 < r2;
        const unsigned mask = (unsigned)__ballot(inball);
        if (mask) {
            if (!haveFirst) { firstIdx = i0 + __ffs(mask) - 1; haveFirst = true; }
            const int rank = __popc(mask & ((1u << lane) - 1u));
            if (inball && (total + rank) < k) out[total + rank] = p;
            total += __popc(mask);
        }
    }
    // fill remaining slots with first in-ball index (or N-1 if ball empty)
    for (int s = total + lane; s < k; s += 32) out[s] = firstIdx;
}

// ---------------------------------------------------------------------------
// 3. Fused gather + 3-layer MLP (bf16 WMMA) + BN + ReLU + k-maxpool.
//    One workgroup = 128 columns (8 centers x 16 or 4 centers x 32).
// ---------------------------------------------------------------------------
struct MlpArgs {
    const float* w[3];
    const float* gamma[3];
    const float* beta[3];
    const float* mean[3];
    const float* var[3];
    int cin[3];
    int cout[3];
    const int* gi;
    const float* featsT;   // (B,N,CF) transposed gather source, or nullptr
    int k;       // neighbors per center (16 or 32)
    int cpt;     // centers per tile = 128 / k
    int chbase;  // output channel offset (0 or 128)
};

__global__ __launch_bounds__(256) void mlp_kernel(const float* __restrict__ xyz,
                                                  const float* __restrict__ feats,
                                                  const float* __restrict__ newxyz,
                                                  float* __restrict__ out_feat,
                                                  MlpArgs a) {
    extern __shared__ char smem[];
    unsigned short* bufA = (unsigned short*)smem;              // 128x128 bf16 (32 KB)
    unsigned short* bufB = (unsigned short*)(smem + 32768);    // 128x64  bf16 (16 KB)
    unsigned short* wbuf = (unsigned short*)(smem + 49152);    // up to 128x96 (24 KB)
    float* bnInv  = (float*)(smem + 49152 + 24576);
    float* bnBias = bnInv + 128;
    int*   pcol   = (int*)(bnBias + 128);                      // 128 point indices
    float* ccoord = (float*)(pcol + 128);                      // cpt*3 center coords

    const int tid = threadIdx.x;
    const int wave = tid >> 5, lane = tid & 31;
    const int half = lane >> 4;
    const int tilesPerBatch = M_ / a.cpt;
    const int b = blockIdx.x / tilesPerBatch;
    const int m0 = (blockIdx.x % tilesPerBatch) * a.cpt;
    const int k = a.k;

    // ---- per-column point index + center coords, once
    if (tid < 128) {
        const int mt = m0 + tid / k;
        pcol[tid] = a.gi[(size_t)(b * M_ + mt) * k + (tid % k)];
    }
    if (tid < a.cpt * 3) ccoord[tid] = newxyz[(size_t)(b * M_ + m0) * 3 + tid];
    __syncthreads();

    // ---- gather X0 (99 real rows, zero-padded to 128) col-major into bufA
    // dword-granular: each element covers a (row, row+1) pair of one column
    unsigned* bufA32 = (unsigned*)bufA;
    for (int e = tid; e < 128 * 64; e += 256) {
        const int col = e >> 6, rr = e & 63;
        const int row0 = rr * 2;
        float v0 = 0.0f, v1 = 0.0f;
        if (row0 < CF_ + 3) {
            const int p = pcol[col];
            const int ci = col / k;
            if (a.featsT) {
                const float* ps = a.featsT + ((size_t)b * N_ + p) * CF_;
                if (row0 < 3) {
                    v0 = xyz[((size_t)b * N_ + p) * 3 + row0] - ccoord[ci * 3 + row0];
                    v1 = (row0 + 1 < 3)
                           ? xyz[((size_t)b * N_ + p) * 3 + row0 + 1] - ccoord[ci * 3 + row0 + 1]
                           : ps[row0 - 2];
                } else {
                    v0 = ps[row0 - 3];
                    if (row0 + 1 < CF_ + 3) v1 = ps[row0 - 2];
                }
            } else {
                if (row0 < 3) {
                    v0 = xyz[((size_t)b * N_ + p) * 3 + row0] - ccoord[ci * 3 + row0];
                    v1 = (row0 + 1 < 3)
                           ? xyz[((size_t)b * N_ + p) * 3 + row0 + 1] - ccoord[ci * 3 + row0 + 1]
                           : feats[((size_t)b * CF_ + 0) * N_ + p];
                } else {
                    v0 = feats[((size_t)b * CF_ + (row0 - 3)) * N_ + p];
                    if (row0 + 1 < CF_ + 3) v1 = feats[((size_t)b * CF_ + (row0 - 2)) * N_ + p];
                }
            }
        }
        bufA32[col * 64 + rr] = pack2bf(v0, v1);
    }
    __syncthreads();

    unsigned short* inb = bufA;
    int ldIn = 128;   // padded K of current layer input

    for (int l = 0; l < 3; ++l) {
        const int cin  = a.cin[l];
        const int cout = a.cout[l];
        const int Kp   = ldIn;

        // cooperative weight load -> bf16 LDS, zero-padded K (only layer 0 pads)
        for (int e = tid; e < cout * Kp; e += 256) {
            const int r = e / Kp, c = e - r * Kp;
            wbuf[e] = (c < cin) ? f2bf(a.w[l][(size_t)r * cin + c]) : (unsigned short)0;
        }
        for (int r = tid; r < cout; r += 256) {
            const float inv = a.gamma[l][r] * rsqrtf(a.var[l][r] + 1e-5f);
            bnInv[r]  = inv;
            bnBias[r] = a.beta[l][r] - a.mean[l][r] * inv;
        }
        __syncthreads();

        if (l < 2) {
            unsigned short* outb = (l == 0) ? bufB : bufA;
            unsigned* outb32 = (unsigned*)outb;
            const int ldOut = cout;
            const int rtN = cout >> 4;
            for (int t = wave; t < rtN * 8; t += 8) {
                const int rt = t >> 3, ct = t & 7;
                v8f acc = {};
                for (int kk = 0; kk < Kp; kk += 32) {
                    const v16bf af = loadA(wbuf, Kp, rt, kk, lane);
                    const v16bf bf = loadB(inb, Kp, ct, kk, lane);
                    acc = __builtin_amdgcn_wmma_f32_16x16x32_bf16(
                        false, af, false, bf, (short)0, acc, false, false);
                }
                const int n = ct * 16 + (lane & 15);
                const int colbase = n * (ldOut >> 1);
#pragma unroll
                for (int r2 = 0; r2 < 4; ++r2) {
                    const int mrow = rt * 16 + half * 8 + 2 * r2;
                    float v0 = acc[2 * r2]     * bnInv[mrow]     + bnBias[mrow];
                    float v1 = acc[2 * r2 + 1] * bnInv[mrow + 1] + bnBias[mrow + 1];
                    v0 = v0 > 0.0f ? v0 : 0.0f;
                    v1 = v1 > 0.0f ? v1 : 0.0f;
                    outb32[colbase + (mrow >> 1)] = pack2bf(v0, v1);
                }
            }
            __syncthreads();
            inb = outb;
            ldIn = ldOut;
        } else {
            // final layer: fuse BN+ReLU+max over k columns per center
            const int rtN = cout >> 4;         // 8
            const int ktiles = k >> 4;         // 1 or 2
            for (int t = wave; t < rtN * a.cpt; t += 8) {
                const int rt = t % rtN, ci = t / rtN;
                float mx[8];
#pragma unroll
                for (int r = 0; r < 8; ++r) mx[r] = -1e30f;
                for (int sub = 0; sub < ktiles; ++sub) {
                    const int ct = ci * ktiles + sub;
                    v8f acc = {};
                    for (int kk = 0; kk < Kp; kk += 32) {
                        const v16bf af = loadA(wbuf, Kp, rt, kk, lane);
                        const v16bf bf = loadB(inb, Kp, ct, kk, lane);
                        acc = __builtin_amdgcn_wmma_f32_16x16x32_bf16(
                            false, af, false, bf, (short)0, acc, false, false);
                    }
#pragma unroll
                    for (int r = 0; r < 8; ++r) {
                        const int mrow = rt * 16 + half * 8 + r;
                        float v = acc[r] * bnInv[mrow] + bnBias[mrow];
                        v = v > 0.0f ? v : 0.0f;
                        mx[r] = fmaxf(mx[r], v);
                    }
                }
                // max across the 16 column-lanes of each half
#pragma unroll
                for (int off = 8; off > 0; off >>= 1)
#pragma unroll
                    for (int r = 0; r < 8; ++r)
                        mx[r] = fmaxf(mx[r], __shfl_xor(mx[r], off, 16));
                if ((lane & 15) == 0) {
                    const int mt = m0 + ci;
#pragma unroll
                    for (int r = 0; r < 8; ++r) {
                        const int ch = a.chbase + rt * 16 + half * 8 + r;
                        out_feat[((size_t)b * COUT_TOTAL + ch) * M_ + mt] = mx[r];
                    }
                }
            }
            __syncthreads();
        }
    }
}

// ---------------------------------------------------------------------------
// Host launcher
// ---------------------------------------------------------------------------
extern "C" void kernel_launch(void* const* d_in, const int* in_sizes, int n_in,
                              void* d_out, int out_size, void* d_ws, size_t ws_size,
                              hipStream_t stream) {
    const float* xyz   = (const float*)d_in[0];
    const float* feats = (const float*)d_in[1];

    // params pytree order: for each layer, dict keys sorted: beta,gamma,mean,var,w
    auto P = [&](int spec, int layer, int which) -> const float* {
        return (const float*)d_in[2 + (spec * 3 + layer) * 5 + which];
    };

    float* out      = (float*)d_out;
    float* out_new  = out;                               // (B,M,3)
    float* out_feat = out + (size_t)B_ * M_ * 3;         // (B,256,M)

    int* ws_idx = (int*)d_ws;                            // B*M
    int* gi0    = ws_idx + B_ * M_;                      // B*M*16
    int* gi1    = gi0 + B_ * M_ * 16;                    // B*M*32
    float* featsT = (float*)(gi1 + B_ * M_ * 32);        // optional (B,N,CF)

    const size_t ws_need_base = (size_t)B_ * M_ * (1 + 16 + 32) * sizeof(int);
    const size_t ws_need_T    = ws_need_base + (size_t)B_ * N_ * CF_ * sizeof(float);
    const bool useT = (ws_size >= ws_need_T);

    // 0. Optional coalescing transpose of features: (B,CF,N) -> (B,N,CF)
    if (useT) {
        transpose_kernel<<<B_ * (CF_ / 32) * (N_ / 32), 256, 0, stream>>>(feats, featsT);
    }

    // 1. FPS
    const size_t fps_smem = (size_t)N_ * 3 * sizeof(float) + 32 * sizeof(float)
                          + 32 * sizeof(int) + sizeof(int);
    fps_kernel<<<B_, 1024, fps_smem, stream>>>(xyz, ws_idx, out);

    // 2. Ball queries
    ballquery_kernel<<<(B_ * M_) / 8, 256, 0, stream>>>(xyz, out_new, gi0, 0.2f * 0.2f, 16);
    ballquery_kernel<<<(B_ * M_) / 8, 256, 0, stream>>>(xyz, out_new, gi1, 0.4f * 0.4f, 32);

    // 3. Fused MLPs
    const size_t mlp_smem = 32768 + 16384 + 24576 + 2 * 128 * sizeof(float)
                          + 128 * sizeof(int) + 32 * sizeof(float);

    MlpArgs a0{};
    MlpArgs a1{};
    const int cin0[3]  = {CF_ + 3, 64, 64},  cout0[3] = {64, 64, 128};
    const int cin1[3]  = {CF_ + 3, 64, 96},  cout1[3] = {64, 96, 128};
    for (int l = 0; l < 3; ++l) {
        a0.beta[l] = P(0, l, 0); a0.gamma[l] = P(0, l, 1); a0.mean[l] = P(0, l, 2);
        a0.var[l]  = P(0, l, 3); a0.w[l]     = P(0, l, 4);
        a0.cin[l] = cin0[l]; a0.cout[l] = cout0[l];
        a1.beta[l] = P(1, l, 0); a1.gamma[l] = P(1, l, 1); a1.mean[l] = P(1, l, 2);
        a1.var[l]  = P(1, l, 3); a1.w[l]     = P(1, l, 4);
        a1.cin[l] = cin1[l]; a1.cout[l] = cout1[l];
    }
    a0.gi = gi0; a0.featsT = useT ? featsT : nullptr; a0.k = 16; a0.cpt = 8; a0.chbase = 0;
    a1.gi = gi1; a1.featsT = useT ? featsT : nullptr; a1.k = 32; a1.cpt = 4; a1.chbase = 128;

    mlp_kernel<<<B_ * (M_ / 8), 256, mlp_smem, stream>>>(xyz, feats, out_new, out_feat, a0);
    mlp_kernel<<<B_ * (M_ / 4), 256, mlp_smem, stream>>>(xyz, feats, out_new, out_feat, a1);
}
